// RBlock_91225105367734
// MI455X (gfx1250) — compile-verified, matrix operand order checked by reading, not run on hardware
//
#include <hip/hip_runtime.h>
#include <hip/hip_bf16.h>
#include <math.h>

typedef __attribute__((ext_vector_type(16))) _Float16     v16h;
typedef __attribute__((ext_vector_type(8)))  float        v8f;
typedef __attribute__((ext_vector_type(4)))  unsigned int u32x4;
typedef __attribute__((ext_vector_type(8)))  int          i32x8;
typedef __attribute__((ext_vector_type(4)))  int          i32x4;

#define Bn   8
#define Cc   64
#define NSP  4096          // H*W
#define CI   32
#define MPOOL 1024         // (H/2)*(W/2)

// ---------------- ws layout (float units) ----------------
#define OFF_U     0              // 65 floats (u[64], abias) -> pad 128
#define OFF_BB    128            // 8*1024
#define OFF_PSI   8320           // 8*1024*32 halves = 131072 floats (32B-aligned)
#define OFF_Y     139392         // 8*4096*32 floats
#define OFF_WY    1187968        // 8*64*4096 floats
#define OFF_ST    3285120        // 128 floats (scale,shift per channel)

// ---------------------------------------------------------
// 1) fold theta through w1:  u[c] = sum_ci w1[ci]*theta_w[ci,c]; abias = sum w1*theta_b
__global__ void prep_kernel(const float* __restrict__ theta_w,
                            const float* __restrict__ theta_b,
                            const float* __restrict__ cp_w,
                            float* __restrict__ u_out) {
    int c = threadIdx.x;   // 64 threads
    float s = 0.f;
    #pragma unroll
    for (int ci = 0; ci < CI; ++ci) s = fmaf(cp_w[ci], theta_w[ci * Cc + c], s);
    u_out[c] = s;
    if (c == 0) {
        float ab = 0.f;
        #pragma unroll
        for (int ci = 0; ci < CI; ++ci) ab = fmaf(cp_w[ci], theta_b[ci], ab);
        u_out[64] = ab;
    }
}

// ---------------------------------------------------------
// 2) phi/psi conv1x1 + maxpool2x2. Emit bb[b,m] = <w2, phi_pool> and psi packed
//    directly into the WMMA 16-bit B-fragment layout:
//    psi_pack[(b*32+kb)*1024 + t*512 + lane*16 + e], lane=(ci&15)+hi*16,
//    hi=(mk>>3)&1, e=(mk&7)|((mk&16)>>1), mk=m&31, kb=m>>5, t=ci>>4.
__global__ void pool_kernel(const float* __restrict__ x,
                            const float* __restrict__ phi_w, const float* __restrict__ phi_b,
                            const float* __restrict__ psi_w, const float* __restrict__ psi_b,
                            const float* __restrict__ cp_w,
                            float* __restrict__ bb_out, _Float16* __restrict__ psi_pack) {
    __shared__ float s_w[CI * Cc];
    int tid = threadIdx.x;
    int gid = blockIdx.x * blockDim.x + tid;   // 0..8191 -> (b,m)
    int b = gid >> 10;
    int m = gid & 1023;
    int ph = m >> 5, pw = m & 31;
    int base = (ph * 2) * 64 + (pw * 2);
    const float* xb = x + (size_t)b * Cc * NSP;

    // ---- phase 1: phi ----
    for (int i = tid; i < CI * Cc; i += blockDim.x) s_w[i] = phi_w[i];
    __syncthreads();
    float vmax[CI];
    #pragma unroll
    for (int ci = 0; ci < CI; ++ci) vmax[ci] = -3.4e38f;
    #pragma unroll
    for (int p = 0; p < 4; ++p) {
        int off = base + (p >> 1) * 64 + (p & 1);
        float acc[CI];
        #pragma unroll
        for (int ci = 0; ci < CI; ++ci) acc[ci] = phi_b[ci];
        for (int c = 0; c < Cc; ++c) {
            float xv = xb[(size_t)c * NSP + off];
            #pragma unroll
            for (int ci = 0; ci < CI; ++ci) acc[ci] = fmaf(s_w[ci * Cc + c], xv, acc[ci]);
        }
        #pragma unroll
        for (int ci = 0; ci < CI; ++ci) vmax[ci] = fmaxf(vmax[ci], acc[ci]);
    }
    float bbv = 0.f;
    #pragma unroll
    for (int ci = 0; ci < CI; ++ci) bbv = fmaf(cp_w[CI + ci], vmax[ci], bbv);
    bb_out[gid] = bbv;
    __syncthreads();

    // ---- phase 2: psi (reuse LDS) ----
    for (int i = tid; i < CI * Cc; i += blockDim.x) s_w[i] = psi_w[i];
    __syncthreads();
    #pragma unroll
    for (int ci = 0; ci < CI; ++ci) vmax[ci] = -3.4e38f;
    #pragma unroll
    for (int p = 0; p < 4; ++p) {
        int off = base + (p >> 1) * 64 + (p & 1);
        float acc[CI];
        #pragma unroll
        for (int ci = 0; ci < CI; ++ci) acc[ci] = psi_b[ci];
        for (int c = 0; c < Cc; ++c) {
            float xv = xb[(size_t)c * NSP + off];
            #pragma unroll
            for (int ci = 0; ci < CI; ++ci) acc[ci] = fmaf(s_w[ci * Cc + c], xv, acc[ci]);
        }
        #pragma unroll
        for (int ci = 0; ci < CI; ++ci) vmax[ci] = fmaxf(vmax[ci], acc[ci]);
    }
    int kb = m >> 5, mk = m & 31;
    int hi = (mk >> 3) & 1;
    int e  = (mk & 7) | ((mk & 16) >> 1);
    _Float16* dst = psi_pack + ((size_t)b * 32 + kb) * 1024;
    #pragma unroll
    for (int ci = 0; ci < CI; ++ci) {
        int t    = ci >> 4;
        int lane = (ci & 15) + hi * 16;
        dst[t * 512 + lane * 16 + e] = (_Float16)vmax[ci];
    }
}

// ---------------------------------------------------------
// 3) main GEMM: y[b,n,k] = (1/1024) * sum_m relu(a[b,n]+bb[b,m]) * psi[b,m,k]
//    A (16x32 f16) generated in registers (rank-1 + relu -> zero A bytes moved);
//    B (whole batch, 64 KB) staged into LDS once per block by the Tensor Data
//    Mover, consumed as ds_load_b128 fragments. 8 waves/block, 16 rows/wave.
__global__ void __launch_bounds__(256)
gemm_kernel(const float* __restrict__ x,
            const float* __restrict__ u,      // [65]
            const float* __restrict__ bb,     // [B*1024]
            const _Float16* __restrict__ psi_pack,
            float* __restrict__ y) {
    // single struct so s_psi sits at LDS offset 0 (descriptor lds_addr = 0)
    __shared__ struct {
        _Float16 psi[32768];   // 64 KB packed B for this batch
        float    bb[MPOOL];    // 4 KB
    } sm;

    int b    = blockIdx.x >> 5;
    int rb   = blockIdx.x & 31;
    int wave = threadIdx.x >> 5;
    int lane = threadIdx.x & 31;
    int lane_lo = lane & 15;
    int hi   = lane >> 4;
    int n0   = rb * 128 + wave * 16;
    int n    = n0 + lane_lo;

    // ---- TDM: DMA this batch's packed psi (64 KB contiguous) into LDS ----
    if (threadIdx.x < 32) {
        unsigned long long ga =
            (unsigned long long)(uintptr_t)(psi_pack + (size_t)b * 32768);
        u32x4 g0;
        g0[0] = 1u;                                        // count=1, load, no gather
        g0[1] = 0u;                                        // lds_addr = 0
        g0[2] = (unsigned)(ga & 0xffffffffull);            // global_addr[31:0]
        g0[3] = (unsigned)((ga >> 32) & 0x01ffffffull)     // global_addr[56:32]
                | (2u << 30);                              // type = 2 (image)
        i32x8 g1;
        g1[0] = (int)(3u << 16);        // data_size = 8 bytes
        g1[1] = (int)(0x2000u << 16);   // tensor_dim0[15:0] = 8192 (bits 79:48)
        g1[2] = (int)(1u << 16);        // tensor_dim1 = 1
        g1[3] = (int)(0x2000u << 16);   // tile_dim0 = 8192
        g1[4] = 1;                      // tile_dim1 = 1, tile_dim2 = 0
        g1[5] = 0x2000;                 // tensor_dim0_stride = 8192
        g1[6] = 0;
        g1[7] = 0;
        i32x4 z4 = {0, 0, 0, 0};
#if defined(__clang_major__) && (__clang_major__ >= 23)
        i32x8 z8 = {0, 0, 0, 0, 0, 0, 0, 0};
        __builtin_amdgcn_tensor_load_to_lds(g0, g1, z4, z4, z8, 0);
#else
        __builtin_amdgcn_tensor_load_to_lds(g0, g1, z4, z4, 0);
#endif
        __builtin_amdgcn_s_wait_tensorcnt(0);
    }
    for (int i = threadIdx.x; i < MPOOL; i += 256) sm.bb[i] = bb[b * MPOOL + i];
    __syncthreads();

    // a[b,n] via folded u (theta never materialized)
    const float* xb = x + (size_t)b * Cc * NSP + n;
    float av = u[64];
    #pragma unroll 8
    for (int c = 0; c < Cc; ++c) av = fmaf(u[c], xb[(size_t)c * NSP], av);

    v8f acc0 = {};
    v8f acc1 = {};

    union fragU { uint4 q[2]; v16h h; };

    for (int kb = 0; kb < 32; ++kb) {
        // A fragment: lanes 0-15 hold K {0..7,16..23}, lanes 16-31 hold K {8..15,24..31}
        int mbase = kb * 32 + hi * 8;
        v16h afrag;
        #pragma unroll
        for (int ee = 0; ee < 8; ++ee)
            afrag[ee] = (_Float16)fmaxf(av + sm.bb[mbase + ee], 0.f);
        #pragma unroll
        for (int ee = 0; ee < 8; ++ee)
            afrag[8 + ee] = (_Float16)fmaxf(av + sm.bb[mbase + 16 + ee], 0.f);

        // B fragments: 32B each, 32B-aligned -> 2x ds_load_b128 apiece
        const uint4* p0 = (const uint4*)(sm.psi + ((size_t)kb * 2 + 0) * 512 + lane * 16);
        const uint4* p1 = (const uint4*)(sm.psi + ((size_t)kb * 2 + 1) * 512 + lane * 16);
        fragU bf0, bf1;
        bf0.q[0] = p0[0];
        bf0.q[1] = p0[1];
        bf1.q[0] = p1[0];
        bf1.q[1] = p1[1];

        acc0 = __builtin_amdgcn_wmma_f32_16x16x32_f16(false, afrag, false, bf0.h,
                                                      (short)0, acc0, false, false);
        acc1 = __builtin_amdgcn_wmma_f32_16x16x32_f16(false, afrag, false, bf1.h,
                                                      (short)0, acc1, false, false);
    }

    const float scale = 1.0f / 1024.0f;
    float* yb = y + (size_t)b * NSP * 32;
    #pragma unroll
    for (int r = 0; r < 8; ++r) {
        int row = n0 + r + hi * 8;            // D layout: vgpr r, lanes16-31 -> M=8+r
        yb[(size_t)row * 32 + lane_lo]      = acc0[r] * scale;
        yb[(size_t)row * 32 + 16 + lane_lo] = acc1[r] * scale;
    }
}

// ---------------------------------------------------------
// 4) wy[b,o,n] = W_b[o] + sum_{k<32} W_w[o,k]*y[b,n,k]  (33rd channel is zero)
__global__ void wconv_kernel(const float* __restrict__ y,
                             const float* __restrict__ W_w,  // [64][33]
                             const float* __restrict__ W_b,
                             float* __restrict__ wy) {
    __shared__ float s_w[Cc * 33];
    __shared__ float s_b[Cc];
    int tid = threadIdx.x;
    for (int i = tid; i < Cc * 33; i += 256) s_w[i] = W_w[i];
    if (tid < Cc) s_b[tid] = W_b[tid];
    __syncthreads();
    int gid = blockIdx.x * 256 + tid;          // (b,n): 32768
    int b = gid >> 12;
    int n = gid & 4095;
    const float* yr = y + (size_t)gid * 32;
    float yv[32];
    #pragma unroll
    for (int k = 0; k < 32; ++k) yv[k] = yr[k];
    float* wyb = wy + (size_t)b * Cc * NSP + n;
    #pragma unroll 4
    for (int o = 0; o < Cc; ++o) {
        float acc = s_b[o];
        #pragma unroll
        for (int k = 0; k < 32; ++k) acc = fmaf(s_w[o * 33 + k], yv[k], acc);
        wyb[(size_t)o * NSP] = acc;            // coalesced over n
    }
}

// ---------------------------------------------------------
// 5) per-channel mean/var over (b,h,w) -> scale/shift
__global__ void stats_kernel(const float* __restrict__ wy,
                             const float* __restrict__ gamma,
                             const float* __restrict__ beta,
                             float* __restrict__ st) {
    int o = blockIdx.x;                         // 64 blocks
    float s = 0.f, s2 = 0.f;
    for (int i = threadIdx.x; i < Bn * NSP; i += 256) {
        int b = i >> 12, n = i & 4095;
        float v = wy[((size_t)b * Cc + o) * NSP + n];
        s  += v;
        s2 += v * v;
    }
    __shared__ float rs[256], rq[256];
    rs[threadIdx.x] = s; rq[threadIdx.x] = s2;
    __syncthreads();
    for (int stp = 128; stp > 0; stp >>= 1) {
        if (threadIdx.x < stp) {
            rs[threadIdx.x] += rs[threadIdx.x + stp];
            rq[threadIdx.x] += rq[threadIdx.x + stp];
        }
        __syncthreads();
    }
    if (threadIdx.x == 0) {
        const float inv = 1.0f / (float)(Bn * NSP);
        float mean = rs[0] * inv;
        float var  = rq[0] * inv - mean * mean;
        float sc   = gamma[o] * rsqrtf(var + 1e-5f);
        st[o * 2]     = sc;
        st[o * 2 + 1] = beta[o] - mean * sc;
    }
}

// ---------------------------------------------------------
// 6) out = wy*scale + shift + x
__global__ void final_kernel(const float* __restrict__ wy,
                             const float* __restrict__ x,
                             const float* __restrict__ st,
                             float* __restrict__ out) {
    size_t gid = (size_t)blockIdx.x * 256 + threadIdx.x;
    int o = (int)((gid >> 12) & 63);
    out[gid] = fmaf(wy[gid], st[o * 2], st[o * 2 + 1] + x[gid]);
}

// ---------------------------------------------------------
extern "C" void kernel_launch(void* const* d_in, const int* in_sizes, int n_in,
                              void* d_out, int out_size, void* d_ws, size_t ws_size,
                              hipStream_t stream) {
    const float* x       = (const float*)d_in[0];
    const float* theta_w = (const float*)d_in[1];
    const float* theta_b = (const float*)d_in[2];
    const float* phi_w   = (const float*)d_in[3];
    const float* phi_b   = (const float*)d_in[4];
    const float* psi_w   = (const float*)d_in[5];
    const float* psi_b   = (const float*)d_in[6];
    const float* cp_w    = (const float*)d_in[7];
    const float* W_w     = (const float*)d_in[8];
    const float* W_b     = (const float*)d_in[9];
    const float* bn_g    = (const float*)d_in[10];
    const float* bn_b    = (const float*)d_in[11];
    float* out = (float*)d_out;

    float* ws = (float*)d_ws;
    float*     w_u   = ws + OFF_U;
    float*     w_bb  = ws + OFF_BB;
    _Float16*  w_psi = (_Float16*)(ws + OFF_PSI);
    float*     w_y   = ws + OFF_Y;
    float*     w_wy  = ws + OFF_WY;
    float*     w_st  = ws + OFF_ST;

    prep_kernel<<<1, 64, 0, stream>>>(theta_w, theta_b, cp_w, w_u);
    pool_kernel<<<32, 256, 0, stream>>>(x, phi_w, phi_b, psi_w, psi_b, cp_w, w_bb, w_psi);
    gemm_kernel<<<256, 256, 0, stream>>>(x, w_u, w_bb, w_psi, w_y);
    wconv_kernel<<<128, 256, 0, stream>>>(w_y, W_w, W_b, w_wy);
    stats_kernel<<<64, 256, 0, stream>>>(w_wy, bn_g, bn_b, w_st);
    final_kernel<<<8192, 256, 0, stream>>>(w_wy, x, w_st, out);
}
